// ROLAND_85856396247964
// MI455X (gfx1250) — matrix operand order repaired and checked by reading, not compiled, not run on hardware
//
#include <hip/hip_runtime.h>
#include <hip/hip_bf16.h>
#include <math.h>

typedef __attribute__((ext_vector_type(16))) __bf16 v16bf;
typedef __attribute__((ext_vector_type(8)))  __bf16 v8bf;
typedef __attribute__((ext_vector_type(8)))  float  v8f;

#define TT 8
#define NN 20000
#define EE 640000
#define DIN 512
#define HH 512
#define DOUT 128
#define ALPHA 0.8f

// ---------------- GEMM: Out[M,Hout] = X[M,K](f32) @ W[K,Hout](f32), bf16 WMMA, f32 accum
// Block: 256 threads (8 waves). Tile M=64, N=128, K-step=32.
// Wave grid 4(M) x 2(N); each wave: 16 rows x 64 cols = 4 wmma tiles sharing one A fragment.
// LDS: A row-major [64][40], B column-major [128][40] -> all fragment traffic is ds_load_b128.
// OOB rows: A pointer clamped to last valid row (unconditional vector loads); stores guarded.
#define TILE_M 64
#define TILE_N 128
#define TILE_K 32
#define LDS_AK 40   // 32 + 8 pad (bf16 elems); row stride 80 B (16B aligned, 20-dword bank stride)
#define LDS_BK 40   // same for column-major B

__global__ __launch_bounds__(256) void gemm_bf16(const float* __restrict__ X,
                                                 const float* __restrict__ W,
                                                 const float* __restrict__ bias,
                                                 float* __restrict__ Out,
                                                 int Mrows, int K, int Hout) {
  __shared__ __bf16 sA[TILE_M * LDS_AK];
  __shared__ __bf16 sBt[TILE_N * LDS_BK];   // [n][k] column-major tile

  const int tid  = threadIdx.x;
  const int lane = tid & 31;
  const int wave = tid >> 5;
  const int wm   = wave & 3;   // 0..3 : M strip
  const int wn   = wave >> 2;  // 0..1 : N strip (64 cols)

  const int rowBase = blockIdx.x * TILE_M;
  const int colBase = blockIdx.y * TILE_N;

  v8f acc[4] = {};

  // A global loads: 64x32 f32 tile, 8 contiguous per thread -> two global_load_b128.
  // Clamp OOB rows to last valid row; results for those rows are never stored.
  const int aRow = tid >> 2;
  const int aCol = (tid & 3) * 8;
  int gRowA = rowBase + aRow;
  if (gRowA >= Mrows) gRowA = Mrows - 1;
  const float* aPtr = X + (size_t)gRowA * (size_t)K + aCol;

  // B global loads: thread owns column n = tid&127, k-group kg = tid>>7 (16 k's).
  // Each of the 16 loads is coalesced across lanes (consecutive n).
  const int bN  = tid & 127;
  const int bKg = tid >> 7;           // 0 or 1 -> k offset 0 / 16
  const float* wColPtr = W + (size_t)(bKg * 16) * (size_t)Hout + colBase + bN;

  const int m16  = lane & 15;
  const int half = lane >> 4;

  // fragment LDS addresses (contiguous 16B chunks per lane)
  const __bf16* aFragBase = &sA[(wm * 16 + m16) * LDS_AK + half * 16];
  const __bf16* bFragBase = &sBt[(wn * 64 + m16) * LDS_BK + half * 16];

  float4 aReg0, aReg1;
  float bReg[16];

  // prologue: global loads for k0 = 0
  aReg0 = *(const float4*)(aPtr);
  aReg1 = *(const float4*)(aPtr + 4);
#pragma unroll
  for (int j = 0; j < 16; ++j) bReg[j] = wColPtr[(size_t)j * Hout];

  for (int k0 = 0; k0 < K; k0 += TILE_K) {
    // convert + store staged registers to LDS (vector stores)
    {
      v8bf ap;
      ap[0] = (__bf16)aReg0.x; ap[1] = (__bf16)aReg0.y;
      ap[2] = (__bf16)aReg0.z; ap[3] = (__bf16)aReg0.w;
      ap[4] = (__bf16)aReg1.x; ap[5] = (__bf16)aReg1.y;
      ap[6] = (__bf16)aReg1.z; ap[7] = (__bf16)aReg1.w;
      *(v8bf*)(&sA[aRow * LDS_AK + aCol]) = ap;

      v8bf bp0, bp1;
#pragma unroll
      for (int j = 0; j < 8; ++j) { bp0[j] = (__bf16)bReg[j]; bp1[j] = (__bf16)bReg[8 + j]; }
      *(v8bf*)(&sBt[bN * LDS_BK + bKg * 16])     = bp0;
      *(v8bf*)(&sBt[bN * LDS_BK + bKg * 16 + 8]) = bp1;
    }
    __syncthreads();

    // issue next tile's global loads (overlap with WMMA below)
    const int kn = k0 + TILE_K;
    if (kn < K) {
      aReg0 = *(const float4*)(aPtr + kn);
      aReg1 = *(const float4*)(aPtr + kn + 4);
      const float* wp = wColPtr + (size_t)kn * Hout;
#pragma unroll
      for (int j = 0; j < 16; ++j) bReg[j] = wp[(size_t)j * Hout];
    }

    // A fragment: two contiguous 16B chunks
    v8bf a0 = *(const v8bf*)(aFragBase);
    v8bf a1 = *(const v8bf*)(aFragBase + 8);
    v16bf afrag = __builtin_shufflevector(a0, a1, 0, 1, 2, 3, 4, 5, 6, 7,
                                          8, 9, 10, 11, 12, 13, 14, 15);

#pragma unroll
    for (int nt = 0; nt < 4; ++nt) {
      const __bf16* bb = bFragBase + nt * 16 * LDS_BK;
      v8bf b0 = *(const v8bf*)(bb);
      v8bf b1 = *(const v8bf*)(bb + 8);
      v16bf bfrag = __builtin_shufflevector(b0, b1, 0, 1, 2, 3, 4, 5, 6, 7,
                                            8, 9, 10, 11, 12, 13, 14, 15);
      acc[nt] = __builtin_amdgcn_wmma_f32_16x16x32_bf16(
          false, afrag, false, bfrag, (short)0, acc[nt], false, false);
    }
    __syncthreads();
  }

  // Epilogue: C/D layout — VGPR v holds row m = v + 8*half, col = lane&15
#pragma unroll
  for (int nt = 0; nt < 4; ++nt) {
    const int n = colBase + wn * 64 + nt * 16 + m16;
    const float bv = bias ? bias[n] : 0.0f;
#pragma unroll
    for (int vv = 0; vv < 8; ++vv) {
      const int m = rowBase + wm * 16 + vv + half * 8;
      if (m < Mrows) Out[(size_t)m * (size_t)Hout + n] = acc[nt][vv] + bv;
    }
  }
}

// ---------------- degree / scatter ----------------
__global__ void deg_init_k(float* deg, int n) {
  int i = blockIdx.x * blockDim.x + threadIdx.x;
  if (i < n) deg[i] = 1.0f;  // self-loop
}
__global__ void deg_acc_k(const int* __restrict__ dst, float* deg, int e) {
  int i = blockIdx.x * blockDim.x + threadIdx.x;
  if (i < e) atomicAdd(&deg[dst[i]], 1.0f);
}
__global__ void deg_fin_k(float* deg, int n) {
  int i = blockIdx.x * blockDim.x + threadIdx.x;
  if (i < n) deg[i] = rsqrtf(deg[i]);
}
// y[i,c] = h[i,c]*dinv[i]^2 + bias[c]  (self-loop term; fully initializes y)
__global__ void scatter_self_k(const float* __restrict__ h, const float* __restrict__ dinv,
                               const float* __restrict__ bias, float* __restrict__ y, int total) {
  int i = blockIdx.x * blockDim.x + threadIdx.x;
  if (i >= total) return;
  int node = i >> 9;
  int c = i & 511;
  float w = dinv[node];
  y[i] = h[i] * w * w + bias[c];
}
#define EPB 8
__global__ __launch_bounds__(256) void scatter_edges_k(const int* __restrict__ src,
                                                       const int* __restrict__ dst,
                                                       const float* __restrict__ h,
                                                       const float* __restrict__ dinv,
                                                       float* __restrict__ y, int e) {
  const int c0 = threadIdx.x * 2;
  const int e0 = blockIdx.x * EPB;
#pragma unroll
  for (int i = 0; i < EPB; ++i) {
    int eIdx = e0 + i;
    if (eIdx >= e) break;
    int s = src[eIdx];
    int d = dst[eIdx];
    float w = dinv[s] * dinv[d];
    const float2 hv = *(const float2*)(h + (size_t)s * 512 + c0);
    float* yp = y + (size_t)d * 512 + c0;
    atomicAdd(yp,     hv.x * w);
    atomicAdd(yp + 1, hv.y * w);
  }
}

// ---------------- LayerNorm + ReLU + EMA (wave per node, H=512) ----------------
__global__ __launch_bounds__(256) void ln_relu_ema_k(const float* __restrict__ Y,
                                                     const float* __restrict__ g,
                                                     const float* __restrict__ beta,
                                                     float* __restrict__ XS,
                                                     const float* __restrict__ XSprev,
                                                     int nrows) {
  const int wave = threadIdx.x >> 5;
  const int lane = threadIdx.x & 31;
  const int node = blockIdx.x * 8 + wave;
  if (node >= nrows) return;
  const float* yp = Y + (size_t)node * 512;
  float v[16];
  float s = 0.0f;
#pragma unroll
  for (int j = 0; j < 16; ++j) { v[j] = yp[j * 32 + lane]; s += v[j]; }
#pragma unroll
  for (int o = 16; o >= 1; o >>= 1) s += __shfl_xor(s, o, 32);
  const float mean = s * (1.0f / 512.0f);
  float vs = 0.0f;
#pragma unroll
  for (int j = 0; j < 16; ++j) { float d = v[j] - mean; vs += d * d; }
#pragma unroll
  for (int o = 16; o >= 1; o >>= 1) vs += __shfl_xor(vs, o, 32);
  const float inv = rsqrtf(vs * (1.0f / 512.0f) + 1e-5f);
  float* xo = XS + (size_t)node * 512;
#pragma unroll
  for (int j = 0; j < 16; ++j) {
    int c = j * 32 + lane;
    float z = (v[j] - mean) * inv * g[c] + beta[c];
    z = fmaxf(z, 0.0f);
    float o = XSprev ? (ALPHA * z + (1.0f - ALPHA) * XSprev[(size_t)node * 512 + c]) : z;
    xo[c] = o;
  }
}

// ---------------- final LN + log_softmax (wave per node, D=128) ----------------
__global__ __launch_bounds__(256) void final_ln_lsm_k(const float* __restrict__ Tm,
                                                      const float* __restrict__ g,
                                                      const float* __restrict__ beta,
                                                      float* __restrict__ out, int nrows) {
  const int wave = threadIdx.x >> 5;
  const int lane = threadIdx.x & 31;
  const int node = blockIdx.x * 8 + wave;
  if (node >= nrows) return;
  const float* tp = Tm + (size_t)node * 128;
  float v[4];
  float s = 0.0f;
#pragma unroll
  for (int j = 0; j < 4; ++j) { v[j] = tp[j * 32 + lane]; s += v[j]; }
#pragma unroll
  for (int o = 16; o >= 1; o >>= 1) s += __shfl_xor(s, o, 32);
  const float mean = s * (1.0f / 128.0f);
  float vs = 0.0f;
#pragma unroll
  for (int j = 0; j < 4; ++j) { float d = v[j] - mean; vs += d * d; }
#pragma unroll
  for (int o = 16; o >= 1; o >>= 1) vs += __shfl_xor(vs, o, 32);
  const float inv = rsqrtf(vs * (1.0f / 128.0f) + 1e-5f);
  float z[4];
  float mx = -INFINITY;
#pragma unroll
  for (int j = 0; j < 4; ++j) {
    int c = j * 32 + lane;
    z[j] = (v[j] - mean) * inv * g[c] + beta[c];
    mx = fmaxf(mx, z[j]);
  }
#pragma unroll
  for (int o = 16; o >= 1; o >>= 1) mx = fmaxf(mx, __shfl_xor(mx, o, 32));
  float es = 0.0f;
#pragma unroll
  for (int j = 0; j < 4; ++j) es += __expf(z[j] - mx);
#pragma unroll
  for (int o = 16; o >= 1; o >>= 1) es += __shfl_xor(es, o, 32);
  const float lse = mx + __logf(es);
  float* op = out + (size_t)node * 128;
#pragma unroll
  for (int j = 0; j < 4; ++j) op[j * 32 + lane] = z[j] - lse;
}

extern "C" void kernel_launch(void* const* d_in, const int* in_sizes, int n_in,
                              void* d_out, int out_size, void* d_ws, size_t ws_size,
                              hipStream_t stream) {
  const float* feats = (const float*)d_in[0];   // [T,N,DIN]
  const int*   adjs  = (const int*)d_in[1];     // [T,2,E]
  const float* W1    = (const float*)d_in[2];
  const float* b1    = (const float*)d_in[3];
  const float* W2    = (const float*)d_in[4];
  const float* b2    = (const float*)d_in[5];
  const float* ln1_g = (const float*)d_in[6];
  const float* ln1_b = (const float*)d_in[7];
  const float* ln2_g = (const float*)d_in[8];
  const float* ln2_b = (const float*)d_in[9];
  const float* lin_W = (const float*)d_in[10];
  const float* lin_b = (const float*)d_in[11];
  const float* lnf_g = (const float*)d_in[12];
  const float* lnf_b = (const float*)d_in[13];

  float* ws   = (float*)d_ws;
  float* xs   = ws;                             // T*N*H
  float* h    = xs + (size_t)TT * NN * HH;      // N*H
  float* y    = h + (size_t)NN * HH;            // N*H
  float* dinv = y + (size_t)NN * HH;            // N (padded)
  float* tmp  = dinv + 20480;                   // N*DOUT

  const dim3 gemmGrid((NN + TILE_M - 1) / TILE_M, HH / TILE_N);
  const int degBlocks = (NN + 255) / 256;
  const int edgBlocks = (EE + 255) / 256;
  const int selfBlocks = (NN * HH + 255) / 256;
  const int scatBlocks = (EE + EPB - 1) / EPB;
  const int nodeBlocks = (NN + 7) / 8;

  for (int layer = 0; layer < 2; ++layer) {
    const float* W  = layer ? W2 : W1;
    const float* bb = layer ? b2 : b1;
    const float* g  = layer ? ln2_g : ln1_g;
    const float* be = layer ? ln2_b : ln1_b;
    for (int t = 0; t < TT; ++t) {
      const float* xin = layer ? (xs + (size_t)t * NN * HH)
                               : (feats + (size_t)t * NN * DIN);
      const int* src = adjs + (size_t)t * 2 * EE;
      const int* dst = src + EE;

      gemm_bf16<<<gemmGrid, 256, 0, stream>>>(xin, W, nullptr, h, NN, DIN, HH);
      deg_init_k<<<degBlocks, 256, 0, stream>>>(dinv, NN);
      deg_acc_k<<<edgBlocks, 256, 0, stream>>>(dst, dinv, EE);
      deg_fin_k<<<degBlocks, 256, 0, stream>>>(dinv, NN);
      scatter_self_k<<<selfBlocks, 256, 0, stream>>>(h, dinv, bb, y, NN * HH);
      scatter_edges_k<<<scatBlocks, 256, 0, stream>>>(src, dst, h, dinv, y, EE);
      const float* prev = (t == 0) ? nullptr : (xs + (size_t)(t - 1) * NN * HH);
      ln_relu_ema_k<<<nodeBlocks, 256, 0, stream>>>(y, g, be,
                                                    xs + (size_t)t * NN * HH, prev, NN);
    }
  }

  // final linear + LN + log_softmax
  const dim3 finGrid((NN + TILE_M - 1) / TILE_M, DOUT / TILE_N);
  gemm_bf16<<<finGrid, 256, 0, stream>>>(xs + (size_t)(TT - 1) * NN * HH, lin_W, lin_b,
                                         tmp, NN, HH, DOUT);
  final_ln_lsm_k<<<nodeBlocks, 256, 0, stream>>>(tmp, lnf_g, lnf_b, (float*)d_out, NN);
}
